// BipartiteGCN_86638080295477
// MI455X (gfx1250) — matrix operand order repaired and checked by reading, not compiled, not run on hardware
//
#include <hip/hip_runtime.h>

typedef __attribute__((ext_vector_type(16))) __bf16 v16bf;
typedef __attribute__((ext_vector_type(8)))  __bf16 v8bf;
typedef __attribute__((ext_vector_type(8)))  float  v8f;

#define SW_STRIDE 136   // 128 + 8 halves pad: 272B row stride -> bank offset 4/col, conflict-free

__device__ __forceinline__ float lrelu_f(float x) { return x > 0.f ? x : 0.01f * x; }

__device__ __forceinline__ void atomAddF(float* p, float v) {
  __hip_atomic_fetch_add(p, v, __ATOMIC_RELAXED, __HIP_MEMORY_SCOPE_AGENT);
}

// A-matrix 16x32 bf16 fragment (ISA 7.12.2 16-bit A layout):
// lane l: row m=l&15, half lh=l>>4; v[0..7]=K koff..koff+7, v[8..15]=K koff+16..koff+23, koff=s*32+lh*8
__device__ __forceinline__ v16bf load_fragA(const __bf16* __restrict__ base, int row, int K, int s, int lh) {
  const __bf16* p = base + (size_t)row * K + s * 32 + lh * 8;
  v8bf lo = *(const v8bf*)(p);
  v8bf hi = *(const v8bf*)(p + 16);
  v16bf r;
#pragma unroll
  for (int i = 0; i < 8; ++i) { r[i] = lo[i]; r[i + 8] = hi[i]; }
  return r;
}

// B-matrix 32x16 bf16 fragment from LDS-staged weights (row stride SW_STRIDE halves):
// lane l holds column n=l&15; v[0..7]=K koff..koff+7, v[8..15]=K koff+16..koff+23, koff=s*32+lh*8
__device__ __forceinline__ v16bf lds_fragB(const __bf16* sw, int col, int s, int lh) {
  const __bf16* p = sw + col * SW_STRIDE + s * 32 + lh * 8;
  v8bf lo = *(const v8bf*)(p);
  v8bf hi = *(const v8bf*)(p + 16);
  v16bf r;
#pragma unroll
  for (int i = 0; i < 8; ++i) { r[i] = lo[i]; r[i + 8] = hi[i]; }
  return r;
}

// cooperative copy of a 128x128 bf16 weight block (columns kb*128.. of a [128][K] row-major
// matrix) into LDS with padded stride; 256 threads, 16B chunks
__device__ __forceinline__ void coop_load_w(__bf16* sw, const __bf16* __restrict__ Wt, int K, int kb) {
  for (int i = threadIdx.x; i < 128 * 16; i += 256) {
    int row = i >> 4, c = i & 15;
    *(uint4*)(sw + row * SW_STRIDE + c * 8) =
        *(const uint4*)(Wt + (size_t)row * K + kb * 128 + c * 8);
  }
}

// ---------------- WMMA GEMM: out[M,128] = act(A[M,K] @ Wt[128,K]^T + bias), K = NKB*128 ----------------
template <int NKB>
__global__ void k_gemm(const __bf16* __restrict__ A, const __bf16* __restrict__ Wt,
                       const float* __restrict__ bias, float* __restrict__ outF,
                       __bf16* __restrict__ outB, int M, int act) {
  constexpr int K = NKB * 128;
  __shared__ __bf16 sw[128 * SW_STRIDE];
  int wid = threadIdx.x >> 5, l = threadIdx.x & 31;
  int lm = l & 15, lh = l >> 4;
  int m0 = (blockIdx.x * 8 + wid) * 16;
  int ra = min(m0 + lm, M - 1);
  v8f acc[8] = {};
#pragma unroll
  for (int kb = 0; kb < NKB; ++kb) {
    __syncthreads();                 // protect sw against readers of previous block
    coop_load_w(sw, Wt, K, kb);
    __syncthreads();
#pragma unroll
    for (int s = 0; s < 4; ++s) {
      v16bf a = load_fragA(A, ra, K, kb * 4 + s, lh);
#pragma unroll
      for (int t = 0; t < 8; ++t) {
        v16bf b = lds_fragB(sw, t * 16 + lm, s, lh);
        acc[t] = __builtin_amdgcn_wmma_f32_16x16x32_bf16(false, a, false, b, (short)0, acc[t], false, false);
      }
    }
  }
  // C/D layout: vgpr d -> row m0 + d + lh*8, col = t*16 + lm
#pragma unroll
  for (int t = 0; t < 8; ++t) {
    int col = t * 16 + lm;
    float bv = bias ? bias[col] : 0.f;
#pragma unroll
    for (int d = 0; d < 8; ++d) {
      int row = m0 + d + lh * 8;
      if (row < M) {
        float x = acc[t][d] + bv;
        if (act) x = lrelu_f(x);
        size_t o = (size_t)row * 128 + col;
        if (outF) outF[o] = x;
        if (outB) outB[o] = (__bf16)x;
      }
    }
  }
}

// ---------------- fused edge kernel: gather -> add -> LN -> LReLU -> WMMA(@wf) -> +bf -> atomic scatter ----
__global__ void k_edge(const float* __restrict__ lp, const float* __restrict__ rp,
                       const int* __restrict__ src, const int* __restrict__ dst,
                       const float* __restrict__ g, const float* __restrict__ bta,
                       const __bf16* __restrict__ Wt, const float* __restrict__ bf,
                       float* __restrict__ agg, int E) {
  __shared__ __bf16 sw[128 * SW_STRIDE];
  coop_load_w(sw, Wt, 128, 0);
  __syncthreads();                    // no further barriers below
  int wid = threadIdx.x >> 5, l = threadIdx.x & 31;
  int e0 = (blockIdx.x * 8 + wid) * 16;
  if (e0 >= E) return;                // wave-uniform
  int lm = l & 15, lh = l >> 4;
  int ee = min(e0 + lm, E - 1);
  const float* lrow = lp + (size_t)src[ee] * 128;
  const float* rrow = rp + (size_t)dst[ee] * 128;

  float x[64];
  float s = 0.f, ss = 0.f;
#pragma unroll
  for (int st = 0; st < 4; ++st) {
#pragma unroll
    for (int half = 0; half < 2; ++half) {
      int k = st * 32 + lh * 8 + half * 16;
      float4 a0 = *(const float4*)(lrow + k);
      float4 a1 = *(const float4*)(lrow + k + 4);
      float4 b0 = *(const float4*)(rrow + k);
      float4 b1 = *(const float4*)(rrow + k + 4);
      float* xp = &x[st * 16 + half * 8];
      xp[0] = a0.x + b0.x; xp[1] = a0.y + b0.y; xp[2] = a0.z + b0.z; xp[3] = a0.w + b0.w;
      xp[4] = a1.x + b1.x; xp[5] = a1.y + b1.y; xp[6] = a1.z + b1.z; xp[7] = a1.w + b1.w;
#pragma unroll
      for (int i = 0; i < 8; ++i) { s += xp[i]; ss += xp[i] * xp[i]; }
    }
  }
  // lanes l and l+16 hold disjoint halves of the same edge-row: combine
  s  += __shfl_xor(s, 16, 32);
  ss += __shfl_xor(ss, 16, 32);
  float mu = s * (1.f / 128.f);
  float var = ss * (1.f / 128.f) - mu * mu;
  float rstd = rsqrtf(var + 1e-5f);

  v8f acc[8] = {};
#pragma unroll
  for (int st = 0; st < 4; ++st) {
    v16bf a;
#pragma unroll
    for (int half = 0; half < 2; ++half) {
      int k = st * 32 + lh * 8 + half * 16;
#pragma unroll
      for (int i = 0; i < 8; ++i) {
        float y = (x[st * 16 + half * 8 + i] - mu) * rstd * g[k + i] + bta[k + i];
        a[half * 8 + i] = (__bf16)lrelu_f(y);
      }
    }
#pragma unroll
    for (int t = 0; t < 8; ++t) {
      v16bf b = lds_fragB(sw, t * 16 + lm, st, lh);
      acc[t] = __builtin_amdgcn_wmma_f32_16x16x32_bf16(false, a, false, b, (short)0, acc[t], false, false);
    }
  }
  // scatter: element (d,t) belongs to edge e0 + d + lh*8, column t*16+lm
#pragma unroll
  for (int d = 0; d < 8; ++d) {
    int er = e0 + d + lh * 8;
    if (er < E) {
      float* arow = agg + (size_t)dst[er] * 128;
#pragma unroll
      for (int t = 0; t < 8; ++t) {
        int col = t * 16 + lm;
        atomAddF(&arow[col], acc[t][d] + bf[col]);
      }
    }
  }
}

// ---------------- per-head: lrelu(v@hw1[h]+hb1[h]) . hw2[h] + hb2[h], masked accumulate ----------------
__global__ void k_head(const __bf16* __restrict__ A, const __bf16* __restrict__ Wt_all,
                       const float* __restrict__ hb1, const float* __restrict__ hw2,
                       const float* __restrict__ hb2, const unsigned char* __restrict__ mask,
                       float* __restrict__ accum, int M) {
  int h = blockIdx.y;
  if (!mask[h]) return;               // block-uniform
  __shared__ __bf16 sw[128 * SW_STRIDE];
  coop_load_w(sw, Wt_all + (size_t)h * 128 * 128, 128, 0);
  __syncthreads();                    // no further barriers below
  int wid = threadIdx.x >> 5, l = threadIdx.x & 31;
  int lm = l & 15, lh = l >> 4;
  int tile = blockIdx.x * 8 + wid;
  int m0 = tile * 16;
  if (m0 >= M) return;                // wave-uniform
  const float* b1 = hb1 + h * 128;
  const float* w2 = hw2 + h * 128;
  int ra = min(m0 + lm, M - 1);
  v8f acc[8] = {};
#pragma unroll
  for (int s = 0; s < 4; ++s) {
    v16bf a = load_fragA(A, ra, 128, s, lh);
#pragma unroll
    for (int t = 0; t < 8; ++t) {
      v16bf b = lds_fragB(sw, t * 16 + lm, s, lh);
      acc[t] = __builtin_amdgcn_wmma_f32_16x16x32_bf16(false, a, false, b, (short)0, acc[t], false, false);
    }
  }
  float p[8] = {};
#pragma unroll
  for (int t = 0; t < 8; ++t) {
    int col = t * 16 + lm;
    float w = w2[col], bb = b1[col];
#pragma unroll
    for (int d = 0; d < 8; ++d) p[d] += lrelu_f(acc[t][d] + bb) * w;
  }
#pragma unroll
  for (int o = 1; o < 16; o <<= 1)
#pragma unroll
    for (int d = 0; d < 8; ++d) p[d] += __shfl_xor(p[d], o, 32);
  if (lm == 0) {
    float b2 = hb2[h];
#pragma unroll
    for (int d = 0; d < 8; ++d) {
      int row = m0 + d + lh * 8;
      if (row < M) atomAddF(&accum[row], p[d] + b2);
    }
  }
}

// ---------------- small-feature embedder: LN(F) -> Linear(F,128) -> LReLU -> bf16 ----------------
__global__ void k_embed(const float* __restrict__ X, int F, const float* __restrict__ g,
                        const float* __restrict__ b, const float* __restrict__ W1,
                        const float* __restrict__ b1, __bf16* __restrict__ out, int N) {
  int row = blockIdx.x;
  int t = threadIdx.x;
  __shared__ float xs[64];
  if (row >= N) { out[(size_t)row * 128 + t] = (__bf16)0.f; return; }
  if (t < F) xs[t] = X[(size_t)row * F + t];
  __syncthreads();
  float mu = 0.f;
  for (int k = 0; k < F; ++k) mu += xs[k];
  mu /= (float)F;
  float vs = 0.f;
  for (int k = 0; k < F; ++k) { float d = xs[k] - mu; vs += d * d; }
  float rstd = rsqrtf(vs / (float)F + 1e-5f);
  float a = b1[t];
  for (int k = 0; k < F; ++k) {
    float y = (xs[k] - mu) * rstd * g[k] + b[k];
    a += y * W1[k * 128 + t];
  }
  out[(size_t)row * 128 + t] = (__bf16)lrelu_f(a);
}

// ---------------- post-conv: mean=agg/max(cnt,1); A2 = [ LN(mean) | right ] as bf16 ----------------
__global__ void k_post(const float* __restrict__ agg, const float* __restrict__ cnt,
                       const float* __restrict__ g, const float* __restrict__ b,
                       const float* __restrict__ right, __bf16* __restrict__ A2, int N) {
  int n = blockIdx.x;
  int t = threadIdx.x; // 128 threads
  __shared__ float red[8];
  float c = fmaxf(cnt[n], 1.f);
  float m = agg[(size_t)n * 128 + t] / c;
  float s = m;
  for (int o = 1; o < 32; o <<= 1) s += __shfl_xor(s, o, 32);
  int wid = t >> 5;
  if ((t & 31) == 0) red[wid] = s;
  __syncthreads();
  float mu = (red[0] + red[1] + red[2] + red[3]) * (1.f / 128.f);
  __syncthreads();
  float d = m - mu;
  float sq = d * d;
  for (int o = 1; o < 32; o <<= 1) sq += __shfl_xor(sq, o, 32);
  if ((t & 31) == 0) red[wid] = sq;
  __syncthreads();
  float rstd = rsqrtf((red[0] + red[1] + red[2] + red[3]) * (1.f / 128.f) + 1e-5f);
  A2[(size_t)n * 256 + t]       = (__bf16)((m - mu) * rstd * g[t] + b[t]);
  A2[(size_t)n * 256 + 128 + t] = (__bf16)right[(size_t)n * 128 + t];
}

// ---------------- helpers ----------------
__global__ void k_zero(float* __restrict__ p, long n) {
  long i = (long)blockIdx.x * blockDim.x + threadIdx.x;
  long st = (long)gridDim.x * blockDim.x;
  for (; i < n; i += st) p[i] = 0.f;
}

__global__ void k_count(const int* __restrict__ dst, float* __restrict__ cnt, int E) {
  int e = blockIdx.x * 256 + threadIdx.x;
  if (e < E) atomAddF(&cnt[dst[e]], 1.f);
}

// W: [nmat][K][128] f32  ->  Wt: [nmat][128][K] bf16
__global__ void k_transw(const float* __restrict__ W, __bf16* __restrict__ Wt, int K, long total) {
  long idx = (long)blockIdx.x * 256 + threadIdx.x;
  if (idx >= total) return;
  long per = (long)K * 128;
  long mat = idx / per;
  long rem = idx - mat * per;
  int k = (int)(rem / 128);
  int n = (int)(rem - (long)k * 128);
  Wt[mat * per + (size_t)n * K + k] = (__bf16)W[idx];
}

__global__ void k_final(const float* __restrict__ accum, const unsigned char* __restrict__ mask,
                        float* __restrict__ out, int N, int NH) {
  int n = blockIdx.x * 256 + threadIdx.x;
  if (n >= N) return;
  float s = 0.f;
  for (int h = 0; h < NH; ++h) s += mask[h] ? 1.f : 0.f;
  out[n] = accum[n] / fmaxf(s, 1.f);
}

extern "C" void kernel_launch(void* const* d_in, const int* in_sizes, int n_in,
                              void* d_out, int out_size, void* d_ws, size_t ws_size,
                              hipStream_t stream) {
  const float* cons_x = (const float*)d_in[0];
  const float* var_x  = (const float*)d_in[1];
  const int*   e_cons = (const int*)d_in[2];
  const int*   e_var  = (const int*)d_in[3];
  const unsigned char* hmask = (const unsigned char*)d_in[4];
  const float* ce_ln_g = (const float*)d_in[5];
  const float* ce_ln_b = (const float*)d_in[6];
  const float* ce_w1 = (const float*)d_in[7];
  const float* ce_b1 = (const float*)d_in[8];
  const float* ce_w2 = (const float*)d_in[9];
  const float* ce_b2 = (const float*)d_in[10];
  const float* ve_ln_g = (const float*)d_in[11];
  const float* ve_ln_b = (const float*)d_in[12];
  const float* ve_w1 = (const float*)d_in[13];
  const float* ve_b1 = (const float*)d_in[14];
  const float* ve_w2 = (const float*)d_in[15];
  const float* ve_b2 = (const float*)d_in[16];
  // vc block (17..29), cv block (30..42): wl,bl,wr,flg,flb,wf,bf,plg,plb,wo1,bo1,wo2,bo2
  const float* vc[13]; for (int i = 0; i < 13; ++i) vc[i] = (const float*)d_in[17 + i];
  const float* cv[13]; for (int i = 0; i < 13; ++i) cv[i] = (const float*)d_in[30 + i];
  const float* hw1 = (const float*)d_in[43];
  const float* hb1 = (const float*)d_in[44];
  const float* hw2 = (const float*)d_in[45];
  const float* hb2 = (const float*)d_in[46];

  const int NC = in_sizes[0] / 5;
  const int NV = in_sizes[1] / 43;
  const int E  = in_sizes[2];
  const int NH = in_sizes[4];
  const int Nmax = NV > NC ? NV : NC;

  // workspace allocator
  char* wp = (char*)d_ws;
  auto alloc = [&](size_t bytes) -> void* {
    void* r = (void*)wp;
    wp += (bytes + 255) & ~(size_t)255;
    return r;
  };
  __bf16* c_emb_bf = (__bf16*)alloc((size_t)NC * 128 * 2);
  __bf16* v_emb_bf = (__bf16*)alloc((size_t)NV * 128 * 2);
  __bf16* c0_bf = (__bf16*)alloc((size_t)NC * 128 * 2);
  float*  c0_f  = (float*)alloc((size_t)NC * 128 * 4);
  __bf16* v0_bf = (__bf16*)alloc((size_t)NV * 128 * 2);
  float*  v0_f  = (float*)alloc((size_t)NV * 128 * 4);
  float*  lp_f  = (float*)alloc((size_t)Nmax * 128 * 4);
  float*  rp_f  = (float*)alloc((size_t)Nmax * 128 * 4);
  float*  agg_f = (float*)alloc((size_t)Nmax * 128 * 4);
  float*  cnt_f = (float*)alloc((size_t)Nmax * 4);
  __bf16* A2_bf = (__bf16*)alloc((size_t)Nmax * 256 * 2);
  __bf16* h1_bf = (__bf16*)alloc((size_t)Nmax * 128 * 2);
  __bf16* c1_bf = (__bf16*)alloc((size_t)NC * 128 * 2);
  __bf16* v1_bf = (__bf16*)alloc((size_t)NV * 128 * 2);
  float*  accum = (float*)alloc((size_t)NV * 4);
  __bf16* wt_ce2 = (__bf16*)alloc(128 * 128 * 2);
  __bf16* wt_ve2 = (__bf16*)alloc(128 * 128 * 2);
  __bf16* wt_vc_wl = (__bf16*)alloc(128 * 128 * 2);
  __bf16* wt_vc_wr = (__bf16*)alloc(128 * 128 * 2);
  __bf16* wt_vc_wf = (__bf16*)alloc(128 * 128 * 2);
  __bf16* wt_vc_wo1 = (__bf16*)alloc(256 * 128 * 2);
  __bf16* wt_vc_wo2 = (__bf16*)alloc(128 * 128 * 2);
  __bf16* wt_cv_wl = (__bf16*)alloc(128 * 128 * 2);
  __bf16* wt_cv_wr = (__bf16*)alloc(128 * 128 * 2);
  __bf16* wt_cv_wf = (__bf16*)alloc(128 * 128 * 2);
  __bf16* wt_cv_wo1 = (__bf16*)alloc(256 * 128 * 2);
  __bf16* wt_cv_wo2 = (__bf16*)alloc(128 * 128 * 2);
  __bf16* wt_hw1 = (__bf16*)alloc((size_t)NH * 128 * 128 * 2);

  auto transw = [&](const float* W, __bf16* Wt, int K, int nmat) {
    long total = (long)nmat * K * 128;
    k_transw<<<dim3((unsigned)((total + 255) / 256)), 256, 0, stream>>>(W, Wt, K, total);
  };
  auto gemm1 = [&](const __bf16* A, const __bf16* Wt, const float* bias,
                   float* outF, __bf16* outB, int M, int act) {
    k_gemm<1><<<dim3((unsigned)((M + 127) / 128)), 256, 0, stream>>>(A, Wt, bias, outF, outB, M, act);
  };
  auto gemm2 = [&](const __bf16* A, const __bf16* Wt, const float* bias,
                   float* outF, __bf16* outB, int M, int act) {
    k_gemm<2><<<dim3((unsigned)((M + 127) / 128)), 256, 0, stream>>>(A, Wt, bias, outF, outB, M, act);
  };
  auto zero = [&](float* p, long n) {
    long b = (n + 255) / 256; if (b > 65535) b = 65535;
    k_zero<<<dim3((unsigned)b), 256, 0, stream>>>(p, n);
  };

  // ---- weight prep ----
  transw(ce_w2, wt_ce2, 128, 1);
  transw(ve_w2, wt_ve2, 128, 1);
  transw(vc[0], wt_vc_wl, 128, 1);
  transw(vc[2], wt_vc_wr, 128, 1);
  transw(vc[5], wt_vc_wf, 128, 1);
  transw(vc[9], wt_vc_wo1, 256, 1);
  transw(vc[11], wt_vc_wo2, 128, 1);
  transw(cv[0], wt_cv_wl, 128, 1);
  transw(cv[2], wt_cv_wr, 128, 1);
  transw(cv[5], wt_cv_wf, 128, 1);
  transw(cv[9], wt_cv_wo1, 256, 1);
  transw(cv[11], wt_cv_wo2, 128, 1);
  transw(hw1, wt_hw1, 128, NH);

  // ---- node embedders ----
  k_embed<<<dim3((unsigned)NC), 128, 0, stream>>>(cons_x, 5, ce_ln_g, ce_ln_b, ce_w1, ce_b1, c_emb_bf, NC);
  k_embed<<<dim3((unsigned)NV), 128, 0, stream>>>(var_x, 43, ve_ln_g, ve_ln_b, ve_w1, ve_b1, v_emb_bf, NV);
  gemm1(c_emb_bf, wt_ce2, ce_b2, c0_f, c0_bf, NC, 1);
  gemm1(v_emb_bf, wt_ve2, ve_b2, v0_f, v0_bf, NV, 1);

  // ---- conv1: v -> c (src=edge_var, dst=edge_cons) ----
  gemm1(v0_bf, wt_vc_wl, vc[1], lp_f, nullptr, NV, 0);          // lp = v@wl+bl
  gemm1(c0_bf, wt_vc_wr, nullptr, rp_f, nullptr, NC, 0);        // rp = c@wr
  zero(agg_f, (long)NC * 128); zero(cnt_f, NC);
  k_count<<<dim3((unsigned)((E + 255) / 256)), 256, 0, stream>>>(e_cons, cnt_f, E);
  k_edge<<<dim3((unsigned)((E + 127) / 128)), 256, 0, stream>>>(lp_f, rp_f, e_var, e_cons,
      vc[3], vc[4], wt_vc_wf, vc[6], agg_f, E);
  k_post<<<dim3((unsigned)NC), 128, 0, stream>>>(agg_f, cnt_f, vc[7], vc[8], c0_f, A2_bf, NC);
  gemm2(A2_bf, wt_vc_wo1, vc[10], nullptr, h1_bf, NC, 1);
  gemm1(h1_bf, wt_vc_wo2, vc[12], nullptr, c1_bf, NC, 0);

  // ---- conv2: c -> v (src=edge_cons, dst=edge_var) ----
  gemm1(c1_bf, wt_cv_wl, cv[1], lp_f, nullptr, NC, 0);
  gemm1(v0_bf, wt_cv_wr, nullptr, rp_f, nullptr, NV, 0);
  zero(agg_f, (long)NV * 128); zero(cnt_f, NV);
  k_count<<<dim3((unsigned)((E + 255) / 256)), 256, 0, stream>>>(e_var, cnt_f, E);
  k_edge<<<dim3((unsigned)((E + 127) / 128)), 256, 0, stream>>>(lp_f, rp_f, e_cons, e_var,
      cv[3], cv[4], wt_cv_wf, cv[6], agg_f, E);
  k_post<<<dim3((unsigned)NV), 128, 0, stream>>>(agg_f, cnt_f, cv[7], cv[8], v0_f, A2_bf, NV);
  gemm2(A2_bf, wt_cv_wo1, cv[10], nullptr, h1_bf, NV, 1);
  gemm1(h1_bf, wt_cv_wo2, cv[12], nullptr, v1_bf, NV, 0);

  // ---- heads ----
  zero(accum, NV);
  k_head<<<dim3((unsigned)(((NV + 15) / 16 + 7) / 8), (unsigned)NH), 256, 0, stream>>>(
      v1_bf, wt_hw1, hb1, hw2, hb2, hmask, accum, NV);
  k_final<<<dim3((unsigned)((NV + 255) / 256)), 256, 0, stream>>>(accum, hmask, (float*)d_out, NV, NH);
}
